// MultiHeadCausalAttention_910533067132
// MI455X (gfx1250) — matrix operand order repaired and checked by reading, not compiled, not run on hardware
//
#include <hip/hip_runtime.h>
#include <hip/hip_bf16.h>

// ---------------- problem constants ----------------
constexpr int T_SEQ  = 2048;
constexpr int NHEADS = 4;
constexpr int DMODEL = 128;
constexpr int DH     = 32;                // head dim
constexpr int BATCH  = 8;
constexpr int MROWS  = BATCH * T_SEQ;     // 16384
// 1/sqrt(32) * log2(e): attention done in base-2 domain so exp2f -> bare v_exp_f32
constexpr float QSCALE2 = 0.17677669529663687f * 1.4426950408889634f;

// ---------------- WMMA types / helpers (CDNA5 / gfx1250) ----------------
typedef __bf16 v16bf __attribute__((ext_vector_type(16)));
typedef __bf16 v8bf  __attribute__((ext_vector_type(8)));
typedef float  v8f   __attribute__((ext_vector_type(8)));

union BF16x16 {
    v16bf          v;
    v8bf           h[2];      // two 16-byte halves: s[0..7], s[8..15]
    unsigned       u32v[8];
    unsigned short s[16];
};

// packed f32x2 -> bf16x2 (single v_cvt_pk_bf16_f32 when available)
#if __has_builtin(__builtin_amdgcn_cvt_pk_bf16_f32)
typedef __bf16 v2bf __attribute__((ext_vector_type(2)));
__device__ __forceinline__ unsigned pack_bf2(float a, float b) {
    union { v2bf v; unsigned u; } x;
    x.v = __builtin_amdgcn_cvt_pk_bf16_f32(a, b);
    return x.u;
}
#else
__device__ __forceinline__ unsigned pack_bf2(float a, float b) {
    union { __bf16 h; unsigned short u; } xa, xb;
    xa.h = (__bf16)a; xb.h = (__bf16)b;
    return (unsigned)xa.u | ((unsigned)xb.u << 16);
}
#endif

__device__ __forceinline__ v8f wmma_bf16(const BF16x16 &a, const BF16x16 &b, v8f c) {
    return __builtin_amdgcn_wmma_f32_16x16x32_bf16(false, a.v, false, b.v,
                                                   (short)0, c, false, false);
}

// A-matrix (16x32 bf16): lane -> M; s[i] <- K = half*16 + i
// B-matrix (32x16 bf16): lane -> N; s[i] <- K = half*16 + i
__device__ __forceinline__ int a_kbase(int j, int half) {
    return (j << 1) + (half << 4);
}

// =====================================================================
// Setup 0: x (f32) -> xbf (bf16), flat copy
// =====================================================================
__global__ __launch_bounds__(256)
void convert_x_kernel(const float* __restrict__ x, __bf16* __restrict__ xbf) {
    const int gid = blockIdx.x * 256 + threadIdx.x;     // 8 elements each
    const float4 a = ((const float4*)x)[gid * 2];
    const float4 b = ((const float4*)x)[gid * 2 + 1];
    uint4 o;
    o.x = pack_bf2(a.x, a.y);
    o.y = pack_bf2(a.z, a.w);
    o.z = pack_bf2(b.x, b.y);
    o.w = pack_bf2(b.z, b.w);
    ((uint4*)xbf)[gid] = o;
}

// =====================================================================
// Setup 1: W [128][N] f32 -> Wt [N][128] bf16 (B-operand friendly)
// =====================================================================
__global__ __launch_bounds__(128)
void transpose_w_kernel(const float* __restrict__ W, __bf16* __restrict__ Wt, int N) {
    const int n = blockIdx.x;
    const int k = threadIdx.x;
    Wt[(size_t)n * DMODEL + k] = (__bf16)W[(size_t)k * N + n];
}

// =====================================================================
// Kernel 1: qkv = xbf @ Wqt^T + b ; Q scaled by (1/sqrt(Dh))*log2(e),
// Q/K -> [B,H,T,32] bf16, V -> [B,H,32,T] bf16 (transposed)
// =====================================================================
__global__ __launch_bounds__(32)
void qkv_proj_kernel(const __bf16* __restrict__ xbf, const __bf16* __restrict__ Wqt,
                     const float* __restrict__ bias,
                     __bf16* __restrict__ Qw, __bf16* __restrict__ Kw,
                     __bf16* __restrict__ Vt) {
    const int lane = threadIdx.x & 31;
    const int half = lane >> 4;
    const int n16  = lane & 15;
    const int m0   = blockIdx.x * 16;
    const int n0   = blockIdx.y * 32;

    const __bf16 *xr  = xbf + (size_t)(m0 + n16) * DMODEL;
    const __bf16 *wr0 = Wqt + (size_t)(n0 + n16) * DMODEL;
    const __bf16 *wr1 = Wqt + (size_t)(n0 + 16 + n16) * DMODEL;

    v8f c0 = {}; v8f c1 = {};
#pragma unroll
    for (int kc = 0; kc < DMODEL; kc += 32) {
        BF16x16 a, b0, b1;
        a.h[0]  = *(const v8bf*)(xr  + kc + half * 16);
        a.h[1]  = *(const v8bf*)(xr  + kc + half * 16 + 8);
        b0.h[0] = *(const v8bf*)(wr0 + kc + half * 16);
        b0.h[1] = *(const v8bf*)(wr0 + kc + half * 16 + 8);
        b1.h[0] = *(const v8bf*)(wr1 + kc + half * 16);
        b1.h[1] = *(const v8bf*)(wr1 + kc + half * 16 + 8);
        c0 = wmma_bf16(a, b0, c0);
        c1 = wmma_bf16(a, b1, c1);
    }

    const float bb0 = bias[n0 + n16];
    const float bb1 = bias[n0 + 16 + n16];
    const int which = n0 / DMODEL;             // 0=Q 1=K 2=V (uniform per block)
    const int h     = (n0 % DMODEL) / DH;      // uniform per block
    const float outScale = (which == 0) ? QSCALE2 : 1.0f;

#pragma unroll
    for (int r = 0; r < 8; ++r) {              // C layout: VGPR r -> M = r + 8*half
        const int m = m0 + r + half * 8;
        const int b = m / T_SEQ, t = m % T_SEQ;
        const int bh = b * NHEADS + h;
        const float v0 = (c0[r] + bb0) * outScale;
        const float v1 = (c1[r] + bb1) * outScale;
        if (which == 2) {                      // V transposed: [bh][d][t]
            Vt[((size_t)bh * DH + n16)      * T_SEQ + t] = (__bf16)v0;
            Vt[((size_t)bh * DH + n16 + 16) * T_SEQ + t] = (__bf16)v1;
        } else {
            __bf16 *dst = ((which == 0) ? Qw : Kw) + ((size_t)bh * T_SEQ + t) * DH;
            dst[n16]      = (__bf16)v0;
            dst[n16 + 16] = (__bf16)v1;
        }
    }
}

// =====================================================================
// Kernel 2: causal flash attention, one wave per 16-query tile,
// 64-key chunks. Softmax in base-2 (exp2f -> bare v_exp_f32).
// Row sums computed on the matrix pipe: P @ ones (no shfl-add tree).
// =====================================================================
__global__ __launch_bounds__(32)
void flash_attn_kernel(const __bf16* __restrict__ Q, const __bf16* __restrict__ K,
                       const __bf16* __restrict__ Vt, __bf16* __restrict__ O) {
    __shared__ __bf16 Plds[16][66];            // 16 x 64 tile, pad 66

    const int lane = threadIdx.x & 31;
    const int half = lane >> 4;
    const int n16  = lane & 15;

    const int ntq  = T_SEQ / 16;
    const int tile = blockIdx.x;
    const int bh   = tile / ntq;
    const int q0   = (tile % ntq) * 16;
    const int b    = bh / NHEADS;
    const int h    = bh % NHEADS;

    const __bf16 *Qb = Q  + (size_t)bh * T_SEQ * DH;
    const __bf16 *Kb = K  + (size_t)bh * T_SEQ * DH;
    const __bf16 *Vb = Vt + (size_t)bh * DH * T_SEQ;   // [d][t]

    // ---- Q tile, A layout ----
    BF16x16 qa;
    {
        const __bf16 *qr = Qb + (size_t)(q0 + n16) * DH;
        qa.h[0] = *(const v8bf*)(qr + half * 16);
        qa.h[1] = *(const v8bf*)(qr + half * 16 + 8);
    }

    // ---- constant all-ones B operand (bf16 1.0 = 0x3F80) ----
    BF16x16 ones;
#pragma unroll
    for (int j = 0; j < 8; ++j) ones.u32v[j] = 0x3F803F80u;

    float mrow[8], lrow[8];
#pragma unroll
    for (int r = 0; r < 8; ++r) { mrow[r] = -3.0e38f; lrow[r] = 0.0f; }
    v8f o0 = {}; v8f o1 = {};

    for (int j0 = 0; j0 <= q0 + 15; j0 += 64) {
        if (j0 + 64 <= q0 + 15)
            __builtin_prefetch(Kb + (size_t)(j0 + 64) * DH, 0, 0);

        // ---- S = Q K^T over 64 keys: 4 B-sets, 4 WMMAs ----
        v8f s0 = {}, s1 = {}, s2 = {}, s3 = {};
        {
            BF16x16 kb0, kb1, kb2, kb3;
            const __bf16 *k0 = Kb + (size_t)(j0 + n16) * DH      + half * 16;
            const __bf16 *k1 = Kb + (size_t)(j0 + 16 + n16) * DH + half * 16;
            const __bf16 *k2 = Kb + (size_t)(j0 + 32 + n16) * DH + half * 16;
            const __bf16 *k3 = Kb + (size_t)(j0 + 48 + n16) * DH + half * 16;
            kb0.h[0] = *(const v8bf*)(k0); kb0.h[1] = *(const v8bf*)(k0 + 8);
            kb1.h[0] = *(const v8bf*)(k1); kb1.h[1] = *(const v8bf*)(k1 + 8);
            kb2.h[0] = *(const v8bf*)(k2); kb2.h[1] = *(const v8bf*)(k2 + 8);
            kb3.h[0] = *(const v8bf*)(k3); kb3.h[1] = *(const v8bf*)(k3 + 8);
            s0 = wmma_bf16(qa, kb0, s0);
            s1 = wmma_bf16(qa, kb1, s1);
            s2 = wmma_bf16(qa, kb2, s2);
            s3 = wmma_bf16(qa, kb3, s3);
        }

        // ---- causal mask + online softmax (base-2), row = r + 8*half ----
#pragma unroll
        for (int r = 0; r < 8; ++r) {
            const int qg  = q0 + r + half * 8;
            const int k0i = j0 + n16;
            float v0 = (k0i      <= qg) ? s0[r] : -3.0e38f;
            float v1 = (k0i + 16 <= qg) ? s1[r] : -3.0e38f;
            float v2 = (k0i + 32 <= qg) ? s2[r] : -3.0e38f;
            float v3 = (k0i + 48 <= qg) ? s3[r] : -3.0e38f;

            float mx = fmaxf(fmaxf(v0, v1), fmaxf(v2, v3));
#pragma unroll
            for (int off = 1; off < 16; off <<= 1)
                mx = fmaxf(mx, __shfl_xor(mx, off, 32));   // within 16-lane half

            const float mn = fmaxf(mrow[r], mx);
            const float sc = exp2f(mrow[r] - mn);
            mrow[r] = mn;
            lrow[r] *= sc;                       // sums added after P@ones below
            o0[r] *= sc;
            o1[r] *= sc;

            const int M = r + half * 8;
            Plds[M][n16]      = (__bf16)exp2f(v0 - mn);
            Plds[M][n16 + 16] = (__bf16)exp2f(v1 - mn);
            Plds[M][n16 + 32] = (__bf16)exp2f(v2 - mn);
            Plds[M][n16 + 48] = (__bf16)exp2f(v3 - mn);
        }
        __syncthreads();

        // ---- gather P in A layout (two 16x32 halves), 16 x u32 DS loads ----
        BF16x16 pa0, pa1;
#pragma unroll
        for (int j = 0; j < 8; ++j) {
            const int kb = a_kbase(j, half);
            pa0.u32v[j] = *(const unsigned*)&Plds[n16][kb];
            pa1.u32v[j] = *(const unsigned*)&Plds[n16][kb + 32];
        }
        __syncthreads();

        // ---- row sums on the matrix pipe: l += P @ ones ----
        v8f sums = {};
        sums = wmma_bf16(pa0, ones, sums);
        sums = wmma_bf16(pa1, ones, sums);
#pragma unroll
        for (int r = 0; r < 8; ++r) lrow[r] += sums[r];

        // ---- O += P V : V transposed, 4 B-sets, 4 WMMAs ----
        {
            BF16x16 vb00, vb01, vb10, vb11;    // [kchunk][nhalf]
            const __bf16 *va = Vb + (size_t)n16 * T_SEQ        + j0 + half * 16;
            const __bf16 *vb = Vb + (size_t)(n16 + 16) * T_SEQ + j0 + half * 16;
            vb00.h[0] = *(const v8bf*)(va);      vb00.h[1] = *(const v8bf*)(va + 8);
            vb01.h[0] = *(const v8bf*)(vb);      vb01.h[1] = *(const v8bf*)(vb + 8);
            vb10.h[0] = *(const v8bf*)(va + 32); vb10.h[1] = *(const v8bf*)(va + 40);
            vb11.h[0] = *(const v8bf*)(vb + 32); vb11.h[1] = *(const v8bf*)(vb + 40);
            o0 = wmma_bf16(pa0, vb00, o0);
            o1 = wmma_bf16(pa0, vb01, o1);
            o0 = wmma_bf16(pa1, vb10, o0);
            o1 = wmma_bf16(pa1, vb11, o1);
        }
    }

    // ---- normalize, write O (bf16, [row][128]) ----
#pragma unroll
    for (int r = 0; r < 8; ++r) {
        const float inv = 1.0f / lrow[r];
        const int t = q0 + r + half * 8;
        __bf16 *p = O + ((size_t)(b * T_SEQ + t)) * DMODEL + h * DH;
        p[n16]      = (__bf16)(o0[r] * inv);
        p[n16 + 16] = (__bf16)(o1[r] * inv);
    }
}

// =====================================================================
// Kernel 3: out = Ow(bf16) @ Wot^T(bf16) + b_out -> f32  (16384x128x128)
// =====================================================================
__global__ __launch_bounds__(32)
void out_proj_kernel(const __bf16* __restrict__ Ow, const __bf16* __restrict__ Wot,
                     const float* __restrict__ bias, float* __restrict__ out) {
    const int lane = threadIdx.x & 31;
    const int half = lane >> 4;
    const int n16  = lane & 15;
    const int m0   = blockIdx.x * 16;
    const int n0   = blockIdx.y * 32;

    const __bf16 *ar  = Ow  + (size_t)(m0 + n16) * DMODEL;
    const __bf16 *wr0 = Wot + (size_t)(n0 + n16) * DMODEL;
    const __bf16 *wr1 = Wot + (size_t)(n0 + 16 + n16) * DMODEL;

    v8f c0 = {}; v8f c1 = {};
#pragma unroll
    for (int kc = 0; kc < DMODEL; kc += 32) {
        BF16x16 a, b0, b1;
        a.h[0]  = *(const v8bf*)(ar  + kc + half * 16);
        a.h[1]  = *(const v8bf*)(ar  + kc + half * 16 + 8);
        b0.h[0] = *(const v8bf*)(wr0 + kc + half * 16);
        b0.h[1] = *(const v8bf*)(wr0 + kc + half * 16 + 8);
        b1.h[0] = *(const v8bf*)(wr1 + kc + half * 16);
        b1.h[1] = *(const v8bf*)(wr1 + kc + half * 16 + 8);
        c0 = wmma_bf16(a, b0, c0);
        c1 = wmma_bf16(a, b1, c1);
    }

    const float bb0 = bias[n0 + n16];
    const float bb1 = bias[n0 + 16 + n16];
#pragma unroll
    for (int r = 0; r < 8; ++r) {
        const int m = m0 + r + half * 8;
        out[(size_t)m * DMODEL + n0 + n16]      = c0[r] + bb0;
        out[(size_t)m * DMODEL + n0 + 16 + n16] = c1[r] + bb1;
    }
}

// =====================================================================
extern "C" void kernel_launch(void* const* d_in, const int* in_sizes, int n_in,
                              void* d_out, int out_size, void* d_ws, size_t ws_size,
                              hipStream_t stream) {
    (void)in_sizes; (void)n_in; (void)out_size; (void)ws_size;

    const float* x     = (const float*)d_in[0];
    const float* W_qkv = (const float*)d_in[1];
    const float* b_qkv = (const float*)d_in[2];
    const float* W_out = (const float*)d_in[3];
    const float* b_out = (const float*)d_in[4];
    float* out = (float*)d_out;

    // workspace (all bf16, ~21 MB total)
    constexpr size_t NX = (size_t)MROWS * DMODEL;                 // 2,097,152
    __bf16* xbf = (__bf16*)d_ws;
    __bf16* Wqt = xbf + NX;
    __bf16* Wot = Wqt + (size_t)384 * DMODEL;
    __bf16* Qw  = Wot + (size_t)DMODEL * DMODEL;
    __bf16* Kw  = Qw + NX;
    __bf16* Vt  = Kw + NX;
    __bf16* Ow  = Vt + NX;

    convert_x_kernel<<<dim3(NX / (256 * 8)), 256, 0, stream>>>(x, xbf);
    transpose_w_kernel<<<dim3(384), 128, 0, stream>>>(W_qkv, Wqt, 384);
    transpose_w_kernel<<<dim3(DMODEL), 128, 0, stream>>>(W_out, Wot, DMODEL);

    qkv_proj_kernel<<<dim3(MROWS / 16, 384 / 32), 32, 0, stream>>>(
        xbf, Wqt, b_qkv, Qw, Kw, Vt);

    flash_attn_kernel<<<dim3(BATCH * NHEADS * (T_SEQ / 16)), 32, 0, stream>>>(
        Qw, Kw, Vt, Ow);

    out_proj_kernel<<<dim3(MROWS / 16, DMODEL / 32), 32, 0, stream>>>(
        Ow, Wot, b_out, out);
}
